// ModuleCorrelation_82910048682528
// MI455X (gfx1250) — matrix operand order repaired and checked by reading, not compiled, not run on hardware
//
#include <hip/hip_runtime.h>
#include <hip/hip_bf16.h>

typedef _Float16 half_t;
typedef __attribute__((ext_vector_type(16))) _Float16 v16h;
typedef __attribute__((ext_vector_type(8)))  _Float16 v8h;
typedef __attribute__((ext_vector_type(8)))  float    v8f;

#define BB   8
#define CC   128
#define HH   96
#define WW   160
#define DISP 4
#define NCH  81          // 9*9 output channels
#define CPAD 136         // 128 channels padded to 136 halfs (272B = 17*16B, 16B aligned, bank-stagger)
#define NWAVE 18         // 9 dy * 2 ntiles
#define NTHR (NWAVE * 32)

// One workgroup computes one (b, y, 16-wide x tile) of the 81-channel cost volume.
// Wave w: dy = w/2 (0..8 -> displacement dy-4), t = w%2 (N-tile of the 24-px second window).
__global__ __launch_bounds__(NTHR) void corr_wmma_kernel(
    const float* __restrict__ first,
    const float* __restrict__ second,
    float* __restrict__ out)
{
    __shared__ __align__(16) _Float16 Slds[9][24][CPAD];   // [dy row][window px][channel]
    __shared__ __align__(16) _Float16 Flds[16][CPAD];      // [tile px][channel]

    const int x0 = blockIdx.x * 16;
    const int y  = blockIdx.y;
    const int b  = blockIdx.z;
    const int tid = threadIdx.x;

    // ---- Stage first tile: 16 px * 128 ch, f32 -> f16 ----
    for (int i = tid; i < 16 * CC; i += NTHR) {
        int px = i & 15;
        int c  = i >> 4;
        int x  = x0 + px;
        float v = (x < WW) ? first[((b * CC + c) * HH + y) * WW + x] : 0.0f;
        Flds[px][c] = (_Float16)v;
    }

    // ---- Stage second window: 9 rows * 24 px * 128 ch, zero-padded ----
    for (int i = tid; i < 9 * CC * 24; i += NTHR) {
        int px  = i % 24;
        int rc  = i / 24;
        int c   = rc % CC;
        int row = rc / CC;
        int ys  = y  - DISP + row;
        int xs  = x0 - DISP + px;
        float v = (ys >= 0 && ys < HH && xs >= 0 && xs < WW)
                    ? second[((b * CC + c) * HH + ys) * WW + xs] : 0.0f;
        Slds[row][px][c] = (_Float16)v;
    }

    __syncthreads();

    const int wave = tid >> 5;        // 0..17
    const int lane = tid & 31;
    const int dy   = wave >> 1;       // 0..8  (actual displacement dy-4)
    const int t    = wave & 1;        // N-tile: window px 16*t .. 16*t+15
    const int half = lane >> 4;       // 0: lanes 0-15, 1: lanes 16-31
    const int ln16 = lane & 15;

    // B-matrix column index (second-window pixel); px >= 24 columns are never
    // stored, clamp so LDS reads stay in the 24-px window.
    int pxB = t * 16 + ln16;
    if (pxB > 23) pxB = 23;

    v8f acc = {};

#pragma unroll
    for (int kk = 0; kk < 4; ++kk) {
        const int kb = kk * 32;
        // A fragment (16x32 f16): lane<16 holds M=lane, K {0..7,16..23};
        //                          lane>=16 holds M=lane-16, K {8..15,24..31}.
        const int m = ln16;
        v8h a_lo = *(const v8h*)&Flds[m][kb + half * 8];
        v8h a_hi = *(const v8h*)&Flds[m][kb + 16 + half * 8];
        v16h a = __builtin_shufflevector(a_lo, a_hi,
                  0,1,2,3,4,5,6,7,8,9,10,11,12,13,14,15);

        // B fragment (32x16 f16): lane<16 holds N=lane, K {0..7,16..23};
        //                          lane>=16 holds N=lane-16, K {8..15,24..31}.
        v8h b_lo = *(const v8h*)&Slds[dy][pxB][kb + half * 8];
        v8h b_hi = *(const v8h*)&Slds[dy][pxB][kb + 16 + half * 8];
        v16h bm = __builtin_shufflevector(b_lo, b_hi,
                  0,1,2,3,4,5,6,7,8,9,10,11,12,13,14,15);

        acc = __builtin_amdgcn_wmma_f32_16x16x32_f16(
                  false, a, false, bm, (short)0, acc, false, false);
    }

    // ---- Scatter valid diagonals: dx = (16t + n) - 4 - m ----
    const float inv_c = 1.0f / (float)CC;
    const int n = ln16;
#pragma unroll
    for (int r = 0; r < 8; ++r) {
        int m  = r + 8 * half;            // C/D layout: VGPR r holds M=r (+8 for hi lanes)
        int dx = n + 16 * t - 4 - m;
        int x  = x0 + m;
        if (dx >= -DISP && dx <= DISP && x < WW) {
            int ch = dy * 9 + (dx + DISP);   // (dy+4)*9 + (dx+4), dy already offset
            out[((b * NCH + ch) * HH + y) * WW + x] = acc[r] * inv_c;
        }
    }
}

extern "C" void kernel_launch(void* const* d_in, const int* in_sizes, int n_in,
                              void* d_out, int out_size, void* d_ws, size_t ws_size,
                              hipStream_t stream) {
    const float* first  = (const float*)d_in[0];
    const float* second = (const float*)d_in[1];
    float* out = (float*)d_out;

    dim3 grid(WW / 16, HH, BB);   // 10 x 96 x 8 workgroups
    dim3 block(NTHR);             // 18 waves of 32
    corr_wmma_kernel<<<grid, block, 0, stream>>>(first, second, out);
}
